// RobustStateSpaceSimulatorAugmented_MANAM_120259084400
// MI455X (gfx1250) — matrix operand order repaired and checked by reading, not compiled, not run on hardware
//
#include <hip/hip_runtime.h>
#include <hip/hip_bf16.h>

// ---------------------------------------------------------------------------
// RobustStateSpaceSimulatorAugmented on MI455X (gfx1250)
//
// Persistent-state scan kernel: 64 workgroups x 256 threads (8 wave32),
// each owns 16 batch rows, iterates t=0..255 with state in LDS.
// All 5 MLPs per step run as WMMA f32_16x16x32_f16 GEMMs:
//   layer1: [16 x Kpad] @ [Kpad x 512] -> tanh -> f16 hid in LDS
//   layer2: [16 x 512]  @ [512 x dout] -> f32 accum via ds_add_f32
// Weights are pre-converted fp32 -> f16 (K padded to mult of 32 with zeros)
// into d_ws by a pre-pass kernel; they stay L2-resident (~753 KB).
// Output snapshots (184 MB total) use non-temporal stores.
// Activation uses the native CDNA5 V_TANH_F32 trans op when available.
// ---------------------------------------------------------------------------

#define B_TOT 1024
#define T_TOT 256
#define HID   512

typedef __attribute__((ext_vector_type(16))) _Float16 v16h;
typedef __attribute__((ext_vector_type(8)))  _Float16 v8h;
typedef __attribute__((ext_vector_type(8)))  float    v8f;

__device__ __forceinline__ float fast_tanh(float x) {
#if __has_builtin(__builtin_amdgcn_tanhf)
  return __builtin_amdgcn_tanhf(x);      // single V_TANH_F32 trans op
#else
  return tanhf(x);
#endif
}

struct Smem {
  alignas(16) _Float16 in[16][128];    // MLP input tile, K padded (stride 128)
  alignas(16) _Float16 hid[16][512];   // layer-1 output (post-tanh, f16)
  float xf[16][64];
  float xp[16][64];
  float dxp[16][64];
  float acc[16][64];                   // layer-2 accumulator (max dout = 64)
  float xb[16][32];
  float xbn[16][32];                   // pre-clip xb_{t+1}
  float mu[16][32];
  float sig[16][32];
  float u[16][16];
  float y[16][16];
  float yacc[16][16];
  float thf[16][8];
  float thh[16][8];
};

__device__ __forceinline__ v16h pack16(v8h lo, v8h hi) {
  v16h r;
#pragma unroll
  for (int i = 0; i < 8; ++i) { r[i] = lo[i]; r[i + 8] = hi[i]; }
  return r;
}

// A fragment (16x32 f16) from a row-major LDS buffer, per the CDNA5 16-bit
// A-matrix VGPR layout: lane (m = lane&15) holds K = {ks..ks+7, ks+16..ks+23},
// ks = 8*(lane>>4) ... realized as two contiguous 16B ds_load_b128.
__device__ __forceinline__ v16h load_a(const _Float16* base) {
  v8h lo = *(const v8h*)base;
  v8h hi = *(const v8h*)(base + 16);
  return pack16(lo, hi);
}

// layer 1: s.in [16][Kpad(<=128), stride 128] @ w1 [Kpad][512] -> tanh -> s.hid
template <int NKT>
__device__ __forceinline__ void mlp_l1(Smem& s,
                                       const _Float16* __restrict__ w1,
                                       const float* __restrict__ b1, int tid) {
  const int lane = tid & 31, wave = tid >> 5;
  const int m = lane & 15, hs = (lane >> 4) & 1;
  v16h a[NKT];
#pragma unroll
  for (int kt = 0; kt < NKT; ++kt)
    a[kt] = load_a(&s.in[m][kt * 32 + hs * 8]);
#pragma unroll
  for (int nt = 0; nt < 4; ++nt) {
    const int n0 = wave * 64 + nt * 16;
    v8f c = {};
#pragma unroll
    for (int kt = 0; kt < NKT; ++kt) {
      // B layout: lane column n0+m, K = {8*hs..8*hs+7, 16+8*hs..16+8*hs+7}
      v16h bf;
#pragma unroll
      for (int i = 0; i < 16; ++i)
        bf[i] = w1[(size_t)(kt * 32 + 8 * hs + (i & 7) + ((i >> 3) << 4)) * HID + n0 + m];
      c = __builtin_amdgcn_wmma_f32_16x16x32_f16(false, a[kt], false, bf,
                                                 (short)0, c, false, false);
    }
    const float bias = b1[n0 + m];
#pragma unroll
    for (int v = 0; v < 8; ++v) {
      float h = fast_tanh(c[v] + bias);
      s.hid[v + 8 * hs][n0 + m] = (_Float16)h;  // D layout: row v+8*hs, col n0+m
    }
  }
}

// layer 2: s.hid [16][512] @ w2 [512][DOUT] -> ds_add_f32 into s.acc
template <int DOUT>
__device__ __forceinline__ void mlp_l2(Smem& s,
                                       const _Float16* __restrict__ w2, int tid) {
  const int lane = tid & 31, wave = tid >> 5;
  const int m = lane & 15, hs = (lane >> 4) & 1;
#pragma unroll
  for (int nt = 0; nt < DOUT / 16; ++nt) {
    const int n0 = nt * 16;
    v8f c = {};
#pragma unroll
    for (int kk = 0; kk < 2; ++kk) {
      const int kt = wave * 2 + kk;  // waves split K: 16 K-tiles over 8 waves
      v16h a = load_a(&s.hid[m][kt * 32 + hs * 8]);
      v16h bf;
#pragma unroll
      for (int i = 0; i < 16; ++i)
        bf[i] = w2[(size_t)(kt * 32 + 8 * hs + (i & 7) + ((i >> 3) << 4)) * DOUT + n0 + m];
      c = __builtin_amdgcn_wmma_f32_16x16x32_f16(false, a, false, bf,
                                                 (short)0, c, false, false);
    }
#pragma unroll
    for (int v = 0; v < 8; ++v)
      atomicAdd(&s.acc[v + 8 * hs][n0 + m], c[v]);
  }
}

__device__ __forceinline__ void zero_acc(Smem& s, int tid) {
  float* a = &s.acc[0][0];
  for (int i = tid; i < 16 * 64; i += 256) a[i] = 0.0f;
}

__global__ void cvt_pad_f16(const float* __restrict__ src,
                            _Float16* __restrict__ dst,
                            int din, int ncols, int total) {
  int i = blockIdx.x * blockDim.x + threadIdx.x;
  if (i >= total) return;
  int k = i / ncols;
  dst[i] = (k < din) ? (_Float16)src[i] : (_Float16)0.0f;
}

__global__ __launch_bounds__(256) void rsss_main(
    const float* __restrict__ xf0, const float* __restrict__ xb0,
    const float* __restrict__ ug, const float* __restrict__ thfg,
    const float* __restrict__ thhg,
    const _Float16* __restrict__ f_w1, const float* __restrict__ f_b1,
    const _Float16* __restrict__ f_w2, const float* __restrict__ f_b2,
    const _Float16* __restrict__ fb_w1, const float* __restrict__ fb_b1,
    const _Float16* __restrict__ fb_w2, const float* __restrict__ fb_b2,
    const _Float16* __restrict__ gb_w1, const float* __restrict__ gb_b1,
    const _Float16* __restrict__ gb_w2, const float* __restrict__ gb_b2,
    const _Float16* __restrict__ h_w1, const float* __restrict__ h_b1,
    const _Float16* __restrict__ h_w2, const float* __restrict__ h_b2,
    const _Float16* __restrict__ hb_w1, const float* __restrict__ hb_b1,
    const _Float16* __restrict__ hb_w2, const float* __restrict__ hb_b2,
    const float* __restrict__ p_xlo, const float* __restrict__ p_xhi,
    const float* __restrict__ p_xblo, const float* __restrict__ p_xbhi,
    const float* __restrict__ p_ylo, const float* __restrict__ p_yhi,
    float* __restrict__ out) {
  __shared__ Smem s;
  const int tid = threadIdx.x;
  const int b0 = blockIdx.x * 16;

  float* out_xf = out;
  float* out_xb = out_xf + (size_t)B_TOT * T_TOT * 64;
  float* out_y  = out_xb + (size_t)B_TOT * T_TOT * 32;
  float* out_xp = out_y  + (size_t)B_TOT * T_TOT * 16;

  const float xlo = *p_xlo, xhi = *p_xhi;
  const float xblo = *p_xblo, xbhi = *p_xbhi;
  const float ylo = *p_ylo, yhi = *p_yhi;

  // ---- initial state -------------------------------------------------------
  for (int i = tid; i < 16 * 64; i += 256) {
    int r = i >> 6, k = i & 63;
    float v = xf0[(size_t)(b0 + r) * 64 + k];
    s.xf[r][k] = v;
    s.xp[r][k] = v;
  }
  for (int i = tid; i < 16 * 32; i += 256) {
    int r = i >> 5, k = i & 31;
    s.xb[r][k] = xb0[(size_t)(b0 + r) * 32 + k];
    s.mu[r][k] = 0.0f;
    s.sig[r][k] = 1.0f;
  }
  for (int i = tid; i < 16 * 8; i += 256) {
    int r = i >> 3, k = i & 7;
    s.thf[r][k] = thfg[(size_t)(b0 + r) * 8 + k];
    s.thh[r][k] = thhg[(size_t)(b0 + r) * 8 + k];
  }
  __syncthreads();

  // ---- y0 = h(x_f_0) + h_b(x_f_0, x_b_0) ----------------------------------
  for (int i = tid; i < 16 * 96; i += 256) {          // h input: [xp, thh] pad 96
    int r = i / 96, k = i - r * 96;
    float v = (k < 64) ? s.xp[r][k] : (k < 72 ? s.thh[r][k - 64] : 0.0f);
    s.in[r][k] = (_Float16)v;
  }
  zero_acc(s, tid);
  __syncthreads();
  mlp_l1<3>(s, h_w1, h_b1, tid);
  __syncthreads();
  mlp_l2<16>(s, h_w2, tid);
  __syncthreads();
  for (int i = tid; i < 16 * 16; i += 256) {
    int r = i >> 4, k = i & 15;
    s.yacc[r][k] = s.acc[r][k] + h_b2[k];
  }
  __syncthreads();
  for (int i = tid; i < 16 * 96; i += 256) {          // hb input: [xf, xb] = 96
    int r = i / 96, k = i - r * 96;
    float v = (k < 64) ? s.xf[r][k] : s.xb[r][k - 64];
    s.in[r][k] = (_Float16)v;
  }
  zero_acc(s, tid);
  __syncthreads();
  mlp_l1<3>(s, hb_w1, hb_b1, tid);
  __syncthreads();
  mlp_l2<16>(s, hb_w2, tid);
  __syncthreads();
  for (int i = tid; i < 16 * 16; i += 256) {
    int r = i >> 4, k = i & 15;
    s.y[r][k] = s.yacc[r][k] + s.acc[r][k] + hb_b2[k];  // y0 (unclipped)
  }
  __syncthreads();

  // ---- time scan -----------------------------------------------------------
#pragma unroll 1
  for (int t = 0; t < T_TOT; ++t) {
    const float tt = (float)t;
    const float t1 = tt + 1.0f, t2 = tt + 2.0f;

    // snapshot carry state at loop entry (streamed, non-temporal)
    for (int i = tid; i < 16 * 64; i += 256) {
      int r = i >> 6, k = i & 63;
      size_t off = ((size_t)(b0 + r) * T_TOT + t) * 64 + k;
      __builtin_nontemporal_store(s.xf[r][k], out_xf + off);
      __builtin_nontemporal_store(s.xp[r][k], out_xp + off);
    }
    for (int i = tid; i < 16 * 32; i += 256) {
      int r = i >> 5, k = i & 31;
      __builtin_nontemporal_store(
          s.xb[r][k], out_xb + ((size_t)(b0 + r) * T_TOT + t) * 32 + k);
    }
    for (int i = tid; i < 16 * 16; i += 256) {
      int r = i >> 4, k = i & 15;
      __builtin_nontemporal_store(
          s.y[r][k], out_y + ((size_t)(b0 + r) * T_TOT + t) * 16 + k);
    }
    for (int i = tid; i < 16 * 16; i += 256) {        // u_t
      int r = i >> 4, k = i & 15;
      s.u[r][k] = ug[((size_t)(b0 + r) * T_TOT + t) * 16 + k];
    }
    __syncthreads();

    // ---- g_b(xp, xb, u): [64+32+16] pad 128 -> 32 --------------------------
    for (int i = tid; i < 16 * 128; i += 256) {
      int r = i >> 7, k = i & 127;
      float v = (k < 64)  ? s.xp[r][k]
              : (k < 96)  ? s.xb[r][k - 64]
              : (k < 112) ? s.u[r][k - 96] : 0.0f;
      s.in[r][k] = (_Float16)v;
    }
    zero_acc(s, tid);
    __syncthreads();
    mlp_l1<4>(s, gb_w1, gb_b1, tid);
    __syncthreads();
    mlp_l2<32>(s, gb_w2, tid);
    __syncthreads();

    // ---- gamma logic -> xb_{t+1} (pre-clip) --------------------------------
    for (int i = tid; i < 16 * 32; i += 256) {
      int r = i >> 5, k = i & 31;
      float gbv = s.acc[r][k] + gb_b2[k];
      float sg = s.sig[r][k];
      float gp = sg;                    // GAMMA_POWER == 1
      float igp = 1.0f / gp;
      float xbmu = s.xb[r][k] - s.mu[r][k];
      float delta = (gbv * (xbmu + 0.5f * gbv * (1.0f + igp))) * (1.0f - igp);
      float eps = t2 * (sg * sg - t2 / t1)
                + xbmu * (xbmu + gbv * (1.0f + igp)
                          + 0.5f * gbv * gbv * (1.0f + igp * igp));
      float d2e2 = delta * delta - eps * eps;
      bool D1 = (eps >= delta) && (delta >= 0.0f);
      bool D2 = (eps < delta) && (delta < 0.0f);
      bool D3 = (delta < eps) && (eps < 0.0f);
      bool D4 = (2.0f * t2 * t2 / t1 * eps >= d2e2) && (d2e2 > 0.0f);
      float gamma = (D1 | D2 | D3 | D4) ? gp : 1.0f;
      s.xbn[r][k] = gbv / gamma;
    }
    __syncthreads();

    // ---- f(xp, u, theta_f): [64+16+8] pad 96 -> 64 -------------------------
    for (int i = tid; i < 16 * 96; i += 256) {
      int r = i / 96, k = i - r * 96;
      float v = (k < 64) ? s.xp[r][k]
              : (k < 80) ? s.u[r][k - 64]
              : (k < 88) ? s.thf[r][k - 80] : 0.0f;
      s.in[r][k] = (_Float16)v;
    }
    zero_acc(s, tid);
    __syncthreads();
    mlp_l1<3>(s, f_w1, f_b1, tid);
    __syncthreads();
    mlp_l2<64>(s, f_w2, tid);
    __syncthreads();
    for (int i = tid; i < 16 * 64; i += 256) {
      int r = i >> 6, k = i & 63;
      s.dxp[r][k] = s.acc[r][k] + f_b2[k];
    }
    __syncthreads();

    // ---- f_b(xp, xb, u): pad 128 -> 64 -------------------------------------
    for (int i = tid; i < 16 * 128; i += 256) {
      int r = i >> 7, k = i & 127;
      float v = (k < 64)  ? s.xp[r][k]
              : (k < 96)  ? s.xb[r][k - 64]
              : (k < 112) ? s.u[r][k - 96] : 0.0f;
      s.in[r][k] = (_Float16)v;
    }
    zero_acc(s, tid);
    __syncthreads();
    mlp_l1<4>(s, fb_w1, fb_b1, tid);
    __syncthreads();
    mlp_l2<64>(s, fb_w2, tid);
    __syncthreads();
    for (int i = tid; i < 16 * 64; i += 256) {   // state advance (pre-clip)
      int r = i >> 6, k = i & 63;
      float dxf = s.dxp[r][k] + s.acc[r][k] + fb_b2[k];
      s.xf[r][k] += dxf;
      s.xp[r][k] += s.dxp[r][k];
    }
    __syncthreads();

    // ---- h(xp_n, theta_h): pad 96 -> 16 ------------------------------------
    for (int i = tid; i < 16 * 96; i += 256) {
      int r = i / 96, k = i - r * 96;
      float v = (k < 64) ? s.xp[r][k] : (k < 72 ? s.thh[r][k - 64] : 0.0f);
      s.in[r][k] = (_Float16)v;
    }
    zero_acc(s, tid);
    __syncthreads();
    mlp_l1<3>(s, h_w1, h_b1, tid);
    __syncthreads();
    mlp_l2<16>(s, h_w2, tid);
    __syncthreads();
    for (int i = tid; i < 16 * 16; i += 256) {
      int r = i >> 4, k = i & 15;
      s.yacc[r][k] = s.acc[r][k] + h_b2[k];
    }
    __syncthreads();

    // ---- h_b(xf_n, xb_n pre-clip): = 96 -> 16 ------------------------------
    for (int i = tid; i < 16 * 96; i += 256) {
      int r = i / 96, k = i - r * 96;
      float v = (k < 64) ? s.xf[r][k] : s.xbn[r][k - 64];
      s.in[r][k] = (_Float16)v;
    }
    zero_acc(s, tid);
    __syncthreads();
    mlp_l1<3>(s, hb_w1, hb_b1, tid);
    __syncthreads();
    mlp_l2<16>(s, hb_w2, tid);
    __syncthreads();

    // ---- projections + mu/sig recurrence -----------------------------------
    for (int i = tid; i < 16 * 16; i += 256) {
      int r = i >> 4, k = i & 15;
      float y = s.yacc[r][k] + s.acc[r][k] + hb_b2[k];
      s.y[r][k] = fminf(fmaxf(y, ylo), yhi);
    }
    for (int i = tid; i < 16 * 64; i += 256) {
      int r = i >> 6, k = i & 63;
      s.xf[r][k] = fminf(fmaxf(s.xf[r][k], xlo), xhi);
      s.xp[r][k] = fminf(fmaxf(s.xp[r][k], xlo), xhi);
    }
    for (int i = tid; i < 16 * 32; i += 256) {
      int r = i >> 5, k = i & 31;
      float xbc = fminf(fmaxf(s.xbn[r][k], xblo), xbhi);
      float it2 = 1.0f / t2;
      float mu_n = t1 * it2 * s.mu[r][k] + it2 * xbc;
      float sg = s.sig[r][k];
      float sig_n = sqrtf(t1 * it2 * sg * sg + it2 * (xbc - mu_n) * (xbc - mu_n));
      s.xb[r][k] = xbc;
      s.mu[r][k] = mu_n;
      s.sig[r][k] = sig_n;
    }
    __syncthreads();
  }
}

extern "C" void kernel_launch(void* const* d_in, const int* in_sizes, int n_in,
                              void* d_out, int out_size, void* d_ws,
                              size_t ws_size, hipStream_t stream) {
  (void)in_sizes; (void)n_in; (void)out_size; (void)ws_size;

  // f16 weight arena in d_ws (K padded to mult. of 32 with zeros)
  _Float16* ws = (_Float16*)d_ws;
  size_t o = 0;
  _Float16* f_w1h  = ws + o; o += (size_t)96  * 512;
  _Float16* fb_w1h = ws + o; o += (size_t)128 * 512;
  _Float16* gb_w1h = ws + o; o += (size_t)128 * 512;
  _Float16* h_w1h  = ws + o; o += (size_t)96  * 512;
  _Float16* hb_w1h = ws + o; o += (size_t)96  * 512;
  _Float16* f_w2h  = ws + o; o += (size_t)512 * 64;
  _Float16* fb_w2h = ws + o; o += (size_t)512 * 64;
  _Float16* gb_w2h = ws + o; o += (size_t)512 * 32;
  _Float16* h_w2h  = ws + o; o += (size_t)512 * 16;
  _Float16* hb_w2h = ws + o; o += (size_t)512 * 16;

  auto cvt = [&](int idx, _Float16* dst, int din, int ncols, int Kpad) {
    int total = Kpad * ncols;
    cvt_pad_f16<<<(total + 255) / 256, 256, 0, stream>>>(
        (const float*)d_in[idx], dst, din, ncols, total);
  };
  cvt(5,  f_w1h, 88,  512, 96);
  cvt(9,  fb_w1h, 112, 512, 128);
  cvt(13, gb_w1h, 112, 512, 128);
  cvt(17, h_w1h, 72,  512, 96);
  cvt(21, hb_w1h, 96,  512, 96);
  cvt(7,  f_w2h, 512, 64, 512);
  cvt(11, fb_w2h, 512, 64, 512);
  cvt(15, gb_w2h, 512, 32, 512);
  cvt(19, h_w2h, 512, 16, 512);
  cvt(23, hb_w2h, 512, 16, 512);

  rsss_main<<<dim3(B_TOT / 16), dim3(256), 0, stream>>>(
      (const float*)d_in[0], (const float*)d_in[1], (const float*)d_in[2],
      (const float*)d_in[3], (const float*)d_in[4],
      f_w1h, (const float*)d_in[6], f_w2h, (const float*)d_in[8],
      fb_w1h, (const float*)d_in[10], fb_w2h, (const float*)d_in[12],
      gb_w1h, (const float*)d_in[14], gb_w2h, (const float*)d_in[16],
      h_w1h, (const float*)d_in[18], h_w2h, (const float*)d_in[20],
      hb_w1h, (const float*)d_in[22], hb_w2h, (const float*)d_in[24],
      (const float*)d_in[25], (const float*)d_in[26], (const float*)d_in[27],
      (const float*)d_in[28], (const float*)d_in[29], (const float*)d_in[30],
      (float*)d_out);
}
